// MultiHeadAttention_13477607375358
// MI455X (gfx1250) — matrix operand order repaired and checked by reading, not compiled, not run on hardware
//
#include <hip/hip_runtime.h>
#include <hip/hip_bf16.h>

typedef __bf16 bf16_t;
typedef __attribute__((ext_vector_type(16))) __bf16 v16bf;
typedef __attribute__((ext_vector_type(8)))  float  v8f;

struct __attribute__((aligned(16))) BF8 { bf16_t h[8]; };
struct __attribute__((aligned(16))) F4  { float  f[4]; };

#define NBATCH 4
#define SEQ    1024
#define EMB    1024
#define NH     16
#define DH     64
#define NPERS  16
#define KLEN   1040   // SEQ + NPERS
#define KPAD   1056   // padded to multiple of 32
#define SCALEF 0.03125f  // 1/sqrt(1024)

__device__ inline v8f vzero8() {
    v8f c;
#pragma unroll
    for (int i = 0; i < 8; ++i) c[i] = 0.f;
    return c;
}

__device__ inline v8f wmma_bf16(v16bf a, v16bf b, v8f c) {
    return __builtin_amdgcn_wmma_f32_16x16x32_bf16(false, a, false, b, (short)0, c,
                                                   false, false);
}

// A-fragment (16x32 bf16, M x K): element e0..7 -> k = kbase + half*8 + e,
// e8..15 -> k = kbase + 16 + half*8 + (e-8). Row m = lane&15, k contiguous in memory.
__device__ inline v16bf load_a_bf(const bf16_t* base, int ld, int kbase) {
    int lane = threadIdx.x & 31;
    int m = lane & 15, half = lane >> 4;
    const bf16_t* row = base + (size_t)m * ld + kbase + half * 8;
    BF8 lo = *(const BF8*)(row);
    BF8 hi = *(const BF8*)(row + 16);
    v16bf a;
#pragma unroll
    for (int e = 0; e < 8; ++e) { a[e] = lo.h[e]; a[e + 8] = hi.h[e]; }
    return a;
}

// A-fragment from f32 source (convert to bf16 on the fly)
__device__ inline v16bf load_a_f32(const float* base, int ld, int kbase) {
    int lane = threadIdx.x & 31;
    int m = lane & 15, half = lane >> 4;
    const float* row = base + (size_t)m * ld + kbase + half * 8;
    v16bf a;
#pragma unroll
    for (int q = 0; q < 2; ++q) {
        F4 v0 = *(const F4*)(row + 4 * q);
        F4 v1 = *(const F4*)(row + 16 + 4 * q);
#pragma unroll
        for (int j = 0; j < 4; ++j) {
            a[4 * q + j]     = (bf16_t)v0.f[j];
            a[8 + 4 * q + j] = (bf16_t)v1.f[j];
        }
    }
    return a;
}

// B-fragment (32x16, K x N): lane column n = lane&15, e0..7 -> k = half*8+e, e8..15 -> k = 16+half*8+(e-8).
// base indexed so that element (k,n) lives at base[n*ld + k] (k contiguous).
__device__ inline v16bf load_b_f32(const float* base, int ld) {
    int lane = threadIdx.x & 31;
    int nn = lane & 15, half = lane >> 4;
    const float* col = base + (size_t)nn * ld + half * 8;
    v16bf b;
#pragma unroll
    for (int q = 0; q < 4; ++q) {
        F4 v = *(const F4*)(col + (q & 1) * 4 + (q >> 1) * 16);
#pragma unroll
        for (int j = 0; j < 4; ++j) b[4 * q + j] = (bf16_t)v.f[j];
    }
    return b;
}

// B-fragment where the per-lane row pointer (already including column n) is given;
// elements e0..7 -> p[half*8 + e], e8..15 -> p[16 + half*8 + (e-8)].
__device__ inline v16bf load_b_contig(const bf16_t* p) {
    int half = (threadIdx.x >> 4) & 1;
    const bf16_t* q = p + half * 8;
    BF8 lo = *(const BF8*)(q);
    BF8 hi = *(const BF8*)(q + 16);
    v16bf b;
#pragma unroll
    for (int e = 0; e < 8; ++e) { b[e] = lo.h[e]; b[e + 8] = hi.h[e]; }
    return b;
}

// cross-lane reductions within each 16-lane half (wave32)
__device__ inline float hred_max(float v) {
    v = fmaxf(v, __shfl_xor(v, 1, 32));
    v = fmaxf(v, __shfl_xor(v, 2, 32));
    v = fmaxf(v, __shfl_xor(v, 4, 32));
    v = fmaxf(v, __shfl_xor(v, 8, 32));
    return v;
}
__device__ inline float hred_sum(float v) {
    v += __shfl_xor(v, 1, 32);
    v += __shfl_xor(v, 2, 32);
    v += __shfl_xor(v, 4, 32);
    v += __shfl_xor(v, 8, 32);
    return v;
}

// ---------------------------------------------------------------------------
// Kernel 1: per-head 64x64 projections, y = x @ W.T, output bf16.
// blockIdx.y: 0=V, 1=K, 2=Q.  V stored transposed [n][h][d][k] for PV WMMA.
// ---------------------------------------------------------------------------
__global__ void proj_kernel(const float* __restrict__ xv, const float* __restrict__ xk,
                            const float* __restrict__ xq,
                            const float* __restrict__ Wv, const float* __restrict__ Wk,
                            const float* __restrict__ Wq,
                            bf16_t* __restrict__ qb, bf16_t* __restrict__ kb,
                            bf16_t* __restrict__ vt) {
    int which = blockIdx.y;  // 0=v,1=k,2=q
    const float* x = which == 0 ? xv : (which == 1 ? xk : xq);
    const float* W = which == 0 ? Wv : (which == 1 ? Wk : Wq);
    int wave = (blockIdx.x * blockDim.x + threadIdx.x) >> 5;
    int h  = wave >> 8;          // 256 row-tiles per head
    int rt = wave & 255;
    int n  = rt >> 6;            // 64 row-tiles per batch
    int l0 = (rt & 63) * 16;
    int lane = threadIdx.x & 31;
    int half = lane >> 4, nn = lane & 15;

    const float* xbase = x + ((size_t)n * SEQ + l0) * EMB + h * DH;
    v16bf a0 = load_a_f32(xbase, EMB, 0);
    v16bf a1 = load_a_f32(xbase, EMB, 32);

#pragma unroll
    for (int ct = 0; ct < 4; ++ct) {
        v16bf b0 = load_b_f32(W + (size_t)ct * 16 * DH, DH);       // d 0..31
        v16bf b1 = load_b_f32(W + (size_t)ct * 16 * DH + 32, DH);  // d 32..63
        v8f c = vzero8();
        c = wmma_bf16(a0, b0, c);
        c = wmma_bf16(a1, b1, c);
        int col = ct * 16 + nn;
#pragma unroll
        for (int r = 0; r < 8; ++r) {
            int row = r + 8 * half;
            bf16_t val = (bf16_t)c[r];
            if (which == 2)
                qb[(((size_t)n * NH + h) * SEQ + l0 + row) * DH + col] = val;
            else if (which == 1)
                kb[(((size_t)n * NH + h) * KPAD + l0 + row) * DH + col] = val;
            else
                vt[(((size_t)n * NH + h) * DH + col) * KPAD + (l0 + row)] = val;
        }
    }
}

// ---------------------------------------------------------------------------
// Kernel 2: append raw persistent K/V rows (broadcast over n,h) + zero padding.
// ---------------------------------------------------------------------------
__global__ void fill_pers(const float* __restrict__ pk, const float* __restrict__ pv,
                          bf16_t* __restrict__ kb, bf16_t* __restrict__ vt) {
    int idx = blockIdx.x * blockDim.x + threadIdx.x;   // 4*16*32*64 = 131072
    int d  = idx & 63;
    int p  = (idx >> 6) & 31;
    int h  = (idx >> 11) & 15;
    int n  = idx >> 15;
    if (n >= NBATCH) return;
    float kv = (p < NPERS) ? pk[p * DH + d] : 0.f;
    float vv = (p < NPERS) ? pv[p * DH + d] : 0.f;
    kb[(((size_t)n * NH + h) * KPAD + SEQ + p) * DH + d] = (bf16_t)kv;
    vt[(((size_t)n * NH + h) * DH + d) * KPAD + SEQ + p] = (bf16_t)vv;
}

// ---------------------------------------------------------------------------
// Kernel 3: two-pass talking-heads flash attention.
// 16 waves per WG, wave w <-> head w. 16 queries per WG, 32-key tiles.
// LDS: Sl (raw scores all heads) | Pl (normalized probs all heads) | Pm (mixed, per-wave)
// ---------------------------------------------------------------------------
__global__ void attn_kernel(const bf16_t* __restrict__ qb, const bf16_t* __restrict__ kb,
                            const bf16_t* __restrict__ vt, const int* __restrict__ mask,
                            const float* __restrict__ th_pre, const float* __restrict__ th_post,
                            bf16_t* __restrict__ ab) {
    extern __shared__ float lds[];
    float* Sl = lds;                  // [16 heads][16 q][32 k]
    float* Pl = lds + 16 * 16 * 32;   // [16 heads][16 q][32 k]
    float* Pm = lds + 2 * 16 * 16 * 32;  // per-wave mixed probs

    int n  = blockIdx.x >> 6;
    int q0 = (blockIdx.x & 63) * 16;
    int w    = threadIdx.x >> 5;      // head index for this wave
    int lane = threadIdx.x & 31;
    int half = lane >> 4, nn = lane & 15;

    const bf16_t* qbase = qb + (((size_t)n * NH + w) * SEQ + q0) * DH;
    v16bf aQ0 = load_a_bf(qbase, DH, 0);
    v16bf aQ1 = load_a_bf(qbase, DH, 32);

    float thp[16], thq[16], cb = 0.f;
#pragma unroll
    for (int hh = 0; hh < 16; ++hh) {
        thp[hh] = th_pre[w * 16 + hh];
        thq[hh] = th_post[w * 16 + hh];
        cb += thp[hh] * exp2f(-(float)(hh + 1));   // folded ALiBi mix coeff
    }
    const int* mrow = mask + n * SEQ;

    float m_s[8], l_s[8];
#pragma unroll
    for (int r = 0; r < 8; ++r) { m_s[r] = -3.0e38f; l_s[r] = 0.f; }

    const int NT = KPAD / 32;   // 33 tiles

    auto compute_S = [&](int k0) {
#pragma unroll
        for (int s = 0; s < 2; ++s) {
            int key = k0 + s * 16 + nn;
            const bf16_t* krow = kb + (((size_t)n * NH + w) * KPAD + key) * DH;
            v8f c = vzero8();
            c = wmma_bf16(aQ0, load_b_contig(krow), c);
            c = wmma_bf16(aQ1, load_b_contig(krow + 32), c);
#pragma unroll
            for (int r = 0; r < 8; ++r)
                Sl[(w * 16 + r + 8 * half) * 32 + s * 16 + nn] = c[r];
        }
    };

    auto compute_E = [&](int k0, float ev[2][8]) {
#pragma unroll
        for (int s = 0; s < 2; ++s) {
            int col = s * 16 + nn;
            int key = k0 + col;
#pragma unroll
            for (int r = 0; r < 8; ++r) {
                int row = r + 8 * half;
                float e = 0.f;
#pragma unroll
                for (int hh = 0; hh < 16; ++hh)
                    e += thp[hh] * Sl[(hh * 16 + row) * 32 + col];
                if (key < SEQ) {
                    e -= fabsf((float)((q0 + row) - key)) * cb;  // folded ALiBi
                    if (mrow[key] == 0) e = -1e4f;               // post-mix mask
                }
                e *= SCALEF;
                if (key >= KLEN) e = -1e30f;                     // pad columns
                ev[s][r] = e;
            }
        }
    };

    // ---- pass 1: softmax statistics (m, l) per (o=w, q-row) ----
    for (int t = 0; t < NT; ++t) {
        int k0 = t * 32;
        compute_S(k0);
        __syncthreads();
        float ev[2][8];
        compute_E(k0, ev);
#pragma unroll
        for (int r = 0; r < 8; ++r) {
            float tmax = hred_max(fmaxf(ev[0][r], ev[1][r]));
            float mnew = fmaxf(m_s[r], tmax);
            float psum = __expf(ev[0][r] - mnew) + __expf(ev[1][r] - mnew);
            psum = hred_sum(psum);
            l_s[r] = l_s[r] * __expf(m_s[r] - mnew) + psum;
            m_s[r] = mnew;
        }
        __syncthreads();
    }

    float rinv[8];
#pragma unroll
    for (int r = 0; r < 8; ++r) rinv[r] = 1.0f / l_s[r];

    // ---- pass 2: recompute, mix with th_post, accumulate P @ V ----
    v8f acc[4];
#pragma unroll
    for (int dt = 0; dt < 4; ++dt) acc[dt] = vzero8();

    for (int t = 0; t < NT; ++t) {
        int k0 = t * 32;
        compute_S(k0);
        __syncthreads();
        float ev[2][8];
        compute_E(k0, ev);
#pragma unroll
        for (int s = 0; s < 2; ++s)
#pragma unroll
            for (int r = 0; r < 8; ++r)
                Pl[(w * 16 + r + 8 * half) * 32 + s * 16 + nn] =
                    __expf(ev[s][r] - m_s[r]) * rinv[r];
        __syncthreads();

        // post-softmax talking heads: Pmix_w = sum_o th_post[w,o] * P_o
        float pmv[2][8];
#pragma unroll
        for (int s = 0; s < 2; ++s) {
            int col = s * 16 + nn;
#pragma unroll
            for (int r = 0; r < 8; ++r) {
                int row = r + 8 * half;
                float p = 0.f;
#pragma unroll
                for (int o = 0; o < 16; ++o)
                    p += thq[o] * Pl[(o * 16 + row) * 32 + col];
                pmv[s][r] = p;
            }
        }
#pragma unroll
        for (int s = 0; s < 2; ++s)
#pragma unroll
            for (int r = 0; r < 8; ++r)
                Pm[(w * 16 + r + 8 * half) * 32 + s * 16 + nn] = pmv[s][r];

        // A-fragment (16x32 bf16) from this wave's private Pm region
        v16bf pa;
        {
            const float* prow = Pm + (size_t)(w * 16 + nn) * 32;
#pragma unroll
            for (int e = 0; e < 8; ++e) {
                pa[e]     = (bf16_t)prow[half * 8 + e];
                pa[e + 8] = (bf16_t)prow[16 + half * 8 + e];
            }
        }
#pragma unroll
        for (int dt = 0; dt < 4; ++dt) {
            const bf16_t* vrow =
                vt + (((size_t)n * NH + w) * DH + dt * 16 + nn) * KPAD + k0;
            acc[dt] = wmma_bf16(pa, load_b_contig(vrow), acc[dt]);
        }
        __syncthreads();
    }

    // store attention output: [n][q][w*64 + d], bf16
#pragma unroll
    for (int dt = 0; dt < 4; ++dt)
#pragma unroll
        for (int r = 0; r < 8; ++r)
            ab[((size_t)n * SEQ + q0 + r + 8 * half) * EMB + w * DH + dt * 16 + nn] =
                (bf16_t)acc[dt][r];
}

// ---------------------------------------------------------------------------
// Kernel 4: out = attn @ Wo.T + bo  (4096x1024 @ 1024x1024), f32 output.
// One wave per 16x64 output tile.
// ---------------------------------------------------------------------------
__global__ void outproj_kernel(const bf16_t* __restrict__ ab, const float* __restrict__ Wo,
                               const float* __restrict__ bo, float* __restrict__ out) {
    int wave = (blockIdx.x * blockDim.x + threadIdx.x) >> 5;
    int rt = wave >> 4;    // 256 row tiles of 16
    int ct = wave & 15;    // 16 col tiles of 64
    int lane = threadIdx.x & 31;
    int half = lane >> 4, nn = lane & 15;

    const bf16_t* arow = ab + (size_t)rt * 16 * EMB;
    v8f acc[4];
#pragma unroll
    for (int c2 = 0; c2 < 4; ++c2) acc[c2] = vzero8();

    for (int k0 = 0; k0 < EMB; k0 += 32) {
        v16bf a = load_a_bf(arow, EMB, k0);
#pragma unroll
        for (int c2 = 0; c2 < 4; ++c2) {
            const float* wb = Wo + (size_t)(ct * 64 + c2 * 16) * EMB + k0;
            acc[c2] = wmma_bf16(a, load_b_f32(wb, EMB), acc[c2]);
        }
    }
#pragma unroll
    for (int c2 = 0; c2 < 4; ++c2) {
        int o = ct * 64 + c2 * 16 + nn;
        float bias = bo[o];
#pragma unroll
        for (int r = 0; r < 8; ++r)
            out[((size_t)rt * 16 + r + 8 * half) * EMB + o] = acc[c2][r] + bias;
    }
}

// ---------------------------------------------------------------------------
extern "C" void kernel_launch(void* const* d_in, const int* in_sizes, int n_in,
                              void* d_out, int out_size, void* d_ws, size_t ws_size,
                              hipStream_t stream) {
    (void)in_sizes; (void)n_in; (void)out_size; (void)ws_size;
    const float* values   = (const float*)d_in[0];
    const float* keys     = (const float*)d_in[1];
    const float* queries  = (const float*)d_in[2];
    const int*   mask     = (const int*)d_in[3];
    const float* Wv       = (const float*)d_in[4];
    const float* Wk       = (const float*)d_in[5];
    const float* Wq       = (const float*)d_in[6];
    const float* Wo       = (const float*)d_in[7];
    const float* bo       = (const float*)d_in[8];
    const float* th_pre   = (const float*)d_in[9];
    const float* th_post  = (const float*)d_in[10];
    const float* p_keys   = (const float*)d_in[11];
    const float* p_values = (const float*)d_in[12];

    char* w = (char*)d_ws;
    auto walloc = [&](size_t bytes) {
        char* p = w;
        w += (bytes + 255) & ~(size_t)255;
        return p;
    };
    bf16_t* qb = (bf16_t*)walloc((size_t)NBATCH * NH * SEQ * DH * 2);   // ~8.4 MB
    bf16_t* kb = (bf16_t*)walloc((size_t)NBATCH * NH * KPAD * DH * 2);  // ~8.7 MB
    bf16_t* vt = (bf16_t*)walloc((size_t)NBATCH * NH * DH * KPAD * 2);  // ~8.7 MB
    bf16_t* ab = (bf16_t*)walloc((size_t)NBATCH * SEQ * EMB * 2);       // ~8.4 MB

    // 1) QKV per-head projections (bf16 WMMA), 4096 waves per matrix
    proj_kernel<<<dim3(512, 3), 256, 0, stream>>>(values, keys, queries, Wv, Wk, Wq,
                                                  qb, kb, vt);
    // 2) persistent rows + zero pad
    fill_pers<<<512, 256, 0, stream>>>(p_keys, p_values, kb, vt);
    // 3) two-pass talking-heads flash attention (96 KB dynamic LDS)
    attn_kernel<<<NBATCH * (SEQ / 16), 512, 3 * 16 * 16 * 32 * sizeof(float), stream>>>(
        qb, kb, vt, mask, th_pre, th_post, ab);
    // 4) output projection + bias
    outproj_kernel<<<512, 256, 0, stream>>>(ab, Wo, bo, (float*)d_out);
}
